// GraphAttentionLayer_601295422009
// MI455X (gfx1250) — matrix operand order repaired and checked by reading, not compiled, not run on hardware
//
#include <hip/hip_runtime.h>
#include <hip/hip_bf16.h>

// GAT layer: B=8, N=2048, D_IN=256, HID=64
//
// Roofline: adj (134 MB int32) read once ~5.8us @ 23.3 TB/s is the byte floor.
// N^2 exp() is factored away via the piecewise-linearity of leaky_relu:
//   x>0 : exp(x-m)    = c1_i * E1_j,  c1 = exp(f1+f2max-m),  E1 = exp(f2-f2max)
//   x<=0: exp(0.2x-m) = c2_i * E2_j,  c2 = exp(0.2(f1+f2max)-m), E2 = exp(0.2(f2-f2max))
// (all factors <= 1, no overflow; branches agree at x=0 so the select is benign).
// Row sums come free from a 5th WMMA against an all-ones B fragment, which has
// the same C/D lane layout as the output accumulators -> no shuffles/LDS.

#define B_    8
#define N_    2048
#define DIN_  256
#define HID_  64
#define L2E_  1.44269504089f

typedef __attribute__((ext_vector_type(16))) _Float16 v16h;
typedef __attribute__((ext_vector_type(8)))  float    v8f;
typedef __attribute__((ext_vector_type(2)))  _Float16 h2v;

// ---------------------------------------------------------------------------
// Kernel 1: Wh = X @ W via WMMA f16 (f32 accum).
//   One wave owns one 16-row tile of (B*N, HID); loops 4 col-tiles reusing its
//   A fragment. Writes Wh (f32 row-major) and WhhT (f16 [b][h][j]) so kernel
//   4's B fragments load contiguous j-pairs.
// ---------------------------------------------------------------------------
__global__ void __launch_bounds__(256)
k_gemm1(const float* __restrict__ X, const float* __restrict__ W,
        float* __restrict__ Wh, _Float16* __restrict__ WhhT) {
  const int lane  = threadIdx.x & 31;
  const int wave  = threadIdx.x >> 5;
  const int rt    = blockIdx.x * 8 + wave;   // 0..1023 row tiles of 16
  const int row0  = rt * 16;
  const int r     = row0 + (lane & 15);      // A-fragment row for this lane
  const int baseA = (lane & 16) ? 8  : 0;
  const int baseB = (lane & 16) ? 16 : 0;
  const int colN  = lane & 15;

  v8f acc[4] = {};

  for (int k0 = 0; k0 < DIN_; k0 += 32) {
    v16h a;
#pragma unroll
    for (int v = 0; v < 8; ++v) {
      const int k = (v < 4) ? (baseA + 2 * v) : (16 + baseA + 2 * (v - 4));
      const float2 x2 = *(const float2*)(X + (size_t)r * DIN_ + k0 + k);
      a[2 * v]     = (_Float16)x2.x;
      a[2 * v + 1] = (_Float16)x2.y;
    }
#pragma unroll
    for (int ct = 0; ct < 4; ++ct) {
      v16h bf;
#pragma unroll
      for (int e = 0; e < 16; ++e)
        bf[e] = (_Float16)W[(size_t)(k0 + ((e < 8) ? (baseA + e) : (16 + baseA + e - 8))) * HID_ + ct * 16 + colN];
      acc[ct] = __builtin_amdgcn_wmma_f32_16x16x32_f16(
          false, a, false, bf, (short)0, acc[ct], false, false);
    }
  }

  const int b     = row0 / N_;               // 16-row tiles never cross batches
  const int mbase = (lane & 16) ? 8 : 0;
#pragma unroll
  for (int ct = 0; ct < 4; ++ct) {
#pragma unroll
    for (int i = 0; i < 8; ++i) {
      const int rr  = row0 + mbase + i;
      const int h   = ct * 16 + colN;
      const float v = acc[ct][i];
      Wh[(size_t)rr * HID_ + h] = v;
      WhhT[((size_t)b * HID_ + h) * N_ + (rr - b * N_)] = (_Float16)v;
    }
  }
}

// ---------------------------------------------------------------------------
// Kernel 2: f1 = Wh@a1, f2 = Wh@a2, f2max[b] = max_j f2[b,j].
// ---------------------------------------------------------------------------
__global__ void __launch_bounds__(256)
k_fvec(const float* __restrict__ Wh, const float* __restrict__ a,
       float* __restrict__ f1, float* __restrict__ f2,
       float* __restrict__ f2max) {
  __shared__ float red[256];
  const int b = blockIdx.x;
  float lmax = -3.0e38f;
  for (int n = threadIdx.x; n < N_; n += 256) {
    const float* wr = Wh + ((size_t)b * N_ + n) * HID_;
    float s1 = 0.f, s2 = 0.f;
#pragma unroll
    for (int h = 0; h < HID_; ++h) {
      const float w = wr[h];
      s1 += w * a[h];
      s2 += w * a[HID_ + h];
    }
    f1[b * N_ + n] = s1;
    f2[b * N_ + n] = s2;
    lmax = fmaxf(lmax, s2);
  }
  red[threadIdx.x] = lmax;
  __syncthreads();
  for (int s = 128; s > 0; s >>= 1) {
    if (threadIdx.x < s)
      red[threadIdx.x] = fmaxf(red[threadIdx.x], red[threadIdx.x + s]);
    __syncthreads();
  }
  if (threadIdx.x == 0) f2max[b] = red[0];
}

// ---------------------------------------------------------------------------
// Kernel 3: per-j exp table. tab[b][j] = { f2 (f32), E1 (f16), E2 (f16) }.
//   E1 = exp(f2 - f2max) <= 1,  E2 = exp(0.2*(f2 - f2max)) <= 1.
// ---------------------------------------------------------------------------
__global__ void __launch_bounds__(256)
k_tab(const float* __restrict__ f2, const float* __restrict__ f2max,
      float2* __restrict__ tab) {
  const int b = blockIdx.x;
  const float fm = f2max[b];
  for (int n = threadIdx.x; n < N_; n += 256) {
    const float v = f2[b * N_ + n];
    const float d = v - fm;
    h2v e;
    e[0] = (_Float16)__builtin_amdgcn_exp2f(d * L2E_);
    e[1] = (_Float16)__builtin_amdgcn_exp2f(0.2f * d * L2E_);
    float2 t;
    t.x = v;
    t.y = __builtin_bit_cast(float, e);
    tab[b * N_ + n] = t;
  }
}

// ---------------------------------------------------------------------------
// Kernel 4: streaming masked softmax + aggregation, one pass over adj.
//   One wave owns a 16-row i-tile. Per 32-wide j chunk: A fragment
//   p = mask ? (x>0 ? c1*E1 : c2*E2) : 0  (pure f16 muls + selects, no exp),
//   then 5x v_wmma_f32_16x16x32_f16: 4 output tiles + 1 ones-column for the
//   row sums (identical C/D layout -> lane-local normalization at the end).
// ---------------------------------------------------------------------------
__global__ void __launch_bounds__(256)
k_attn(const int* __restrict__ adj, const float* __restrict__ f1,
       const float* __restrict__ f2max, const float2* __restrict__ tab,
       const _Float16* __restrict__ WhhT, float* __restrict__ out) {
  const int lane  = threadIdx.x & 31;
  const int wave  = threadIdx.x >> 5;
  const int t     = blockIdx.x * 8 + wave;   // 0..1023 i-tiles
  const int b     = t >> 7;
  const int i0    = (t & 127) * 16;
  const int irow  = i0 + (lane & 15);
  const int baseA = (lane & 16) ? 8  : 0;
  const int baseB = (lane & 16) ? 16 : 0;
  const int colN  = lane & 15;

  const float f1v = f1[b * N_ + irow];
  const float ub  = f1v + f2max[b];
  const float m   = (ub > 0.f) ? ub : 0.2f * ub;   // lrelu(upper bound)
  const _Float16 c1h = (_Float16)__builtin_amdgcn_exp2f((ub - m) * L2E_);
  const _Float16 c2h = (_Float16)__builtin_amdgcn_exp2f((0.2f * ub - m) * L2E_);
  const _Float16 zh  = (_Float16)0.f;

  const int*      adjRow = adj + ((size_t)b * N_ + irow) * N_;
  const float*    tabB   = (const float*)(tab + (size_t)b * N_);
  const _Float16* WT     = WhhT + (size_t)b * HID_ * N_;

  v16h ones;
#pragma unroll
  for (int e = 0; e < 16; ++e) ones[e] = (_Float16)1.f;

  v8f acc[4] = {};
  v8f sacc   = {};

  for (int j0 = 0; j0 < N_; j0 += 32) {
    if (j0 + 32 < N_) __builtin_prefetch(adjRow + j0 + 32, 0, 1);  // global_prefetch_b8

    v16h p;
#pragma unroll
    for (int v = 0; v < 8; ++v) {
      const int k = (v < 4) ? (baseA + 2 * v) : (16 + baseA + 2 * (v - 4));
      const int j = j0 + k;
      const int2   ad = *(const int2*)(adjRow + j);
      const float4 tt = *(const float4*)(tabB + 2 * j);   // {f2_j, E12_j, f2_j1, E12_j1}
      const h2v e0 = __builtin_bit_cast(h2v, tt.y);
      const h2v e1 = __builtin_bit_cast(h2v, tt.w);
      const float x0 = f1v + tt.x;
      const float x1 = f1v + tt.z;
      _Float16 q0 = (x0 > 0.f) ? (_Float16)(c1h * e0[0]) : (_Float16)(c2h * e0[1]);
      _Float16 q1 = (x1 > 0.f) ? (_Float16)(c1h * e1[0]) : (_Float16)(c2h * e1[1]);
      p[2 * v]     = (ad.x > 0) ? q0 : zh;
      p[2 * v + 1] = (ad.y > 0) ? q1 : zh;
    }
#pragma unroll
    for (int ct = 0; ct < 4; ++ct) {
      v16h bf;
      const _Float16* src = WT + (size_t)(ct * 16 + colN) * N_ + j0;
#pragma unroll
      for (int v = 0; v < 8; ++v) {
        const int k = (v < 4) ? (baseA + 2 * v) : (16 + baseA + 2 * (v - 4));
        bf[2 * v]     = src[k];
        bf[2 * v + 1] = src[k + 1];
      }
      acc[ct] = __builtin_amdgcn_wmma_f32_16x16x32_f16(
          false, p, false, bf, (short)0, acc[ct], false, false);
    }
    // Row sums: S = P @ ones, same C/D layout as acc -> lane-local normalize.
    sacc = __builtin_amdgcn_wmma_f32_16x16x32_f16(
        false, p, false, ones, (short)0, sacc, false, false);
  }

  float rinv[8];
#pragma unroll
  for (int i = 0; i < 8; ++i) rinv[i] = 1.0f / sacc[i];

  const int mbase = (lane & 16) ? 8 : 0;
  float* orow = out + ((size_t)b * N_ + i0) * HID_;
#pragma unroll
  for (int ct = 0; ct < 4; ++ct) {
#pragma unroll
    for (int i = 0; i < 8; ++i)
      orow[(size_t)(mbase + i) * HID_ + ct * 16 + colN] = acc[ct][i] * rinv[i];
  }
}

// ---------------------------------------------------------------------------
// Launcher.  Workspace layout (bytes):
//   [0,       4194304)  Wh    f32  [B*N][HID]
//   [4194304, 6291456)  WhhT  f16  [B][HID][N]
//   [6291456, 6356992)  f1    f32  [B][N]
//   [6356992, 6422528)  f2    f32  [B][N]
//   [6422528, 6422592)  f2max f32  [B]
//   [6422592, 6553664)  tab   8B   [B][N]
// ---------------------------------------------------------------------------
extern "C" void kernel_launch(void* const* d_in, const int* in_sizes, int n_in,
                              void* d_out, int out_size, void* d_ws, size_t ws_size,
                              hipStream_t stream) {
  const float* X   = (const float*)d_in[0];   // node_features (8,2048,256) f32
  const int*   adj = (const int*)d_in[1];     // adj (8,2048,2048) i32
  const float* W   = (const float*)d_in[2];   // W (256,64) f32
  const float* a   = (const float*)d_in[3];   // a (128,1) f32
  float*       out = (float*)d_out;           // (8,2048,64) f32

  char* ws = (char*)d_ws;
  float*     Wh    = (float*)ws;
  _Float16*  WhhT  = (_Float16*)(ws + 4194304);
  float*     f1    = (float*)(ws + 6291456);
  float*     f2    = (float*)(ws + 6356992);
  float*     f2max = (float*)(ws + 6422528);
  float2*    tab   = (float2*)(ws + 6422592);

  k_gemm1<<<128, 256, 0, stream>>>(X, W, Wh, WhhT);
  k_fvec <<<B_,  256, 0, stream>>>(Wh, a, f1, f2, f2max);
  k_tab  <<<B_,  256, 0, stream>>>(f2, f2max, tab);
  k_attn <<<128, 256, 0, stream>>>(adj, f1, f2max, tab, WhhT, out);
}
